// bce_50483045597972
// MI455X (gfx1250) — compile-verified, matrix operand order checked by reading, not run on hardware
//
#include <hip/hip_runtime.h>
#include <math.h>

typedef float   f32x4 __attribute__((ext_vector_type(4)));
typedef int     i32x4 __attribute__((ext_vector_type(4)));
typedef __bf16  v16bf __attribute__((ext_vector_type(16)));
typedef float   v8f   __attribute__((ext_vector_type(8)));

// Raw hardware transcendentals: v_exp_f32 computes exp2, v_log_f32 computes
// log2, ~1 ulp, no libm guard code. Inputs are provably in safe ranges:
// exp2 arg <= 0, log2 arg in [1,2]. Gate on __AMDGCN__ so the host pass
// (where the amdgcn builtins don't exist) falls back to standard libm names.
#if defined(__AMDGCN__) && __has_builtin(__builtin_amdgcn_exp2f) && \
    __has_builtin(__builtin_amdgcn_logf)
#define FAST_EXP2(x) __builtin_amdgcn_exp2f(x)
#define FAST_LOG2(x) __builtin_amdgcn_logf(x)
#else
#define FAST_EXP2(x) exp2f(x)
#define FAST_LOG2(x) log2f(x)
#endif

#define LOG2E_F 1.44269504088896340736f
#define LN2_F   0.69314718055994530942f

// per-element stable BCE term: softplus(t==0 ? x : -x)
//   = ln2 * log2(1 + exp2(-|x| * log2e)) + max(t==0 ? x : -x, 0)
__device__ __forceinline__ float bce_term(float x, int t) {
  float e   = FAST_EXP2(__builtin_fabsf(x) * -LOG2E_F);  // e^{-|x|}, arg <= 0
  float l2  = FAST_LOG2(1.0f + e);                       // arg in [1,2]
  float lin = fmaxf((t == 0) ? x : -x, 0.0f);
  return fmaf(l2, LN2_F, lin);
}

// ---------------------------------------------------------------------------
// Kernel 1: streaming softplus + WMMA-accumulated reduction to per-block sums
// Each wave processes 512-element tiles: 32 lanes x 16 elements. The 16
// per-lane values become the A matrix (16x32 bf16); B = ones; C (f32 16x16)
// accumulates row sums across all tiles.  wave_sum = sum(C over lanes)/16.
// ---------------------------------------------------------------------------
__global__ __launch_bounds__(256) void bce_partial_kernel(
    const float* __restrict__ logits, const int* __restrict__ target,
    float* __restrict__ partials, long long n) {
  const int lane = threadIdx.x & 31;
  const int wave = threadIdx.x >> 5;

  const long long numTiles   = n >> 9;                       // 512 elems/tile
  const long long wavesTotal = (long long)gridDim.x * 8;     // 8 waves/block
  const long long waveGlobal = (long long)blockIdx.x * 8 + wave;

  v16bf ones;
#pragma unroll
  for (int j = 0; j < 16; ++j) ones[j] = (__bf16)1.0f;

  v8f c = {};  // f32 accumulator matrix, persists across the whole stream

  for (long long tile = waveGlobal; tile < numTiles; tile += wavesTotal) {
    const long long base = tile << 9;
    v16bf a;
#pragma unroll
    for (int i = 0; i < 4; ++i) {
      // instruction i covers a contiguous 512B span across the wave
      const long long off = base + (long long)i * 128 + (long long)lane * 4;
      f32x4 xv = __builtin_nontemporal_load((const f32x4*)(logits + off));
      i32x4 tv = __builtin_nontemporal_load((const i32x4*)(target + off));
#pragma unroll
      for (int j = 0; j < 4; ++j) {
        a[i * 4 + j] = (__bf16)bce_term(xv[j], tv[j]);
      }
    }
    // D[m][n] += sum_k A[m][k]  (row sums, replicated over the 16 columns)
    c = __builtin_amdgcn_wmma_f32_16x16x32_bf16(
        /*neg_a=*/false, a, /*neg_b=*/false, ones,
        /*c_mod=*/(short)0, c, /*reuse_a=*/false, /*reuse_b=*/false);
  }

  // Per-lane sum of the 8 accumulator rows. Every row-sum appears in all 16
  // columns, so the true wave total is (sum over lanes) / 16 (exact: /2^4).
  float s = c[0] + c[1] + c[2] + c[3] + c[4] + c[5] + c[6] + c[7];

  // Scalar tail for n % 512 (none for 8192^2; kept for generality).
  float ts = 0.0f;
  {
    const long long tailBase   = numTiles << 9;
    const long long tid        = (long long)blockIdx.x * 256 + threadIdx.x;
    const long long totThreads = (long long)gridDim.x * 256;
    for (long long idx = tailBase + tid; idx < n; idx += totThreads) {
      ts += bce_term(logits[idx], target[idx]);
    }
  }

  float r = s * 0.0625f + ts;
#pragma unroll
  for (int m = 16; m; m >>= 1) r += __shfl_xor(r, m, 32);

  __shared__ float wsum[8];
  if (lane == 0) wsum[wave] = r;
  __syncthreads();
  if (threadIdx.x == 0) {
    float b = 0.0f;
#pragma unroll
    for (int i = 0; i < 8; ++i) b += wsum[i];
    partials[blockIdx.x] = b;
  }
}

// ---------------------------------------------------------------------------
// Kernel 2: reduce per-block partials, write mean
// ---------------------------------------------------------------------------
__global__ __launch_bounds__(256) void bce_finalize_kernel(
    const float* __restrict__ partials, int nPartials, float invN,
    float* __restrict__ outp) {
  float s = 0.0f;
  for (int i = threadIdx.x; i < nPartials; i += 256) s += partials[i];
#pragma unroll
  for (int m = 16; m; m >>= 1) s += __shfl_xor(s, m, 32);

  __shared__ float wsum[8];
  const int wave = threadIdx.x >> 5;
  const int lane = threadIdx.x & 31;
  if (lane == 0) wsum[wave] = s;
  __syncthreads();
  if (threadIdx.x == 0) {
    float t = 0.0f;
#pragma unroll
    for (int i = 0; i < 8; ++i) t += wsum[i];
    outp[0] = t * invN;
  }
}

extern "C" void kernel_launch(void* const* d_in, const int* in_sizes, int n_in,
                              void* d_out, int out_size, void* d_ws, size_t ws_size,
                              hipStream_t stream) {
  (void)n_in; (void)out_size;
  const float* logits = (const float*)d_in[0];
  const int*   target = (const int*)d_in[1];
  const long long n   = (long long)in_sizes[0];

  int blocks = 2048;  // 16384 waves: fills the machine, 8 tiles per wave
  const size_t maxBlocks = ws_size / sizeof(float);
  if ((size_t)blocks > maxBlocks) blocks = (int)maxBlocks;
  if (blocks < 1) blocks = 1;

  float* partials = (float*)d_ws;
  bce_partial_kernel<<<blocks, 256, 0, stream>>>(logits, target, partials, n);
  bce_finalize_kernel<<<1, 256, 0, stream>>>(partials, blocks,
                                             1.0f / (float)n, (float*)d_out);
}
